// MultiHeadAttention_84052509983469
// MI455X (gfx1250) — matrix operand order repaired and checked by reading, not compiled, not run on hardware
//
#include <hip/hip_runtime.h>
#include <hip/hip_bf16.h>

// ---------------------------------------------------------------------------
// MHA forward for MI455X (gfx1250): bf16 WMMA with f32 accumulation,
// TDM (tensor_load_to_lds) staging of K tiles in the attention loop.
//   B=2, S=2048, D=1024, H=16, HD=64
// ---------------------------------------------------------------------------

typedef __attribute__((ext_vector_type(16))) __bf16 v16bf;
typedef __attribute__((ext_vector_type(8)))  __bf16 v8bf;
typedef __attribute__((ext_vector_type(8)))  float  v8f;
typedef __attribute__((ext_vector_type(4)))  unsigned v4u;
typedef __attribute__((ext_vector_type(8)))  int    v8i32;
typedef __attribute__((ext_vector_type(4)))  int    v4i32;

#define WMMA_BF16(a, b, c)                                                    \
  __builtin_amdgcn_wmma_f32_16x16x32_bf16(false, (a), false, (b), (short)0,   \
                                          (c), false, false)

#define B_   2
#define S_   2048
#define D_   1024
#define H_   16
#define HD_  64

#if defined(__has_builtin)
#if __has_builtin(__builtin_amdgcn_tensor_load_to_lds) &&                     \
    __has_builtin(__builtin_amdgcn_s_wait_tensorcnt)
#define USE_TDM 1
#endif
#endif

// A-fragment (16x32 bf16, M x K): lane row = lane%16.
// lanes 0-15 hold K = 0..15; lanes 16-31 hold K = 16..31.
static __device__ inline v16bf load_a_frag(const __bf16* rowbase, int half) {
  v8bf lo = *reinterpret_cast<const v8bf*>(rowbase + half * 16);
  v8bf hi = *reinterpret_cast<const v8bf*>(rowbase + half * 16 + 8);
  v16bf r;
#pragma unroll
  for (int i = 0; i < 8; ++i) { r[i] = lo[i]; r[i + 8] = hi[i]; }
  return r;
}

#ifdef USE_TDM
// Issue a TDM 2D load: 32 rows x 64 cols of bf16 from a row-major
// [2048 x 64] tensor into LDS at byte offset lds_off. Whole-wave op.
static __device__ inline void tdm_load_k_chunk(const __bf16* gsrc,
                                               unsigned lds_off) {
  unsigned long long ga = (unsigned long long)(size_t)gsrc;
  v4u g0;
  g0[0] = 1u;                                   // count=1, no gather
  g0[1] = lds_off;                              // lds_addr (bytes)
  g0[2] = (unsigned)(ga & 0xffffffffu);         // global_addr[31:0]
  g0[3] = (unsigned)((ga >> 32) & 0x01ffffffu)  // global_addr[56:32]
          | (2u << 30);                         // type = 2 ("image")
  v8i32 g1;
  g1[0] = (int)(1u << 16);                      // data_size = 1 -> 2 bytes
  g1[1] = (int)(64u << 16);                     // tensor_dim0 = 64 (lo bits @48+)
  g1[2] = (int)(2048u << 16);                   // tensor_dim1 = 2048
  g1[3] = (int)(64u << 16);                     // tile_dim0 = 64
  g1[4] = 32;                                   // tile_dim1 = 32
  g1[5] = 64;                                   // tensor_dim0_stride = 64
  g1[6] = 0;
  g1[7] = 0;
  v4i32 z4 = {0, 0, 0, 0};
#if __clang_major__ >= 23
  v8i32 z8 = {0, 0, 0, 0, 0, 0, 0, 0};
  __builtin_amdgcn_tensor_load_to_lds(g0, g1, z4, z4, z8, 0);
#else
  __builtin_amdgcn_tensor_load_to_lds(g0, g1, z4, z4, 0);
#endif
}
#endif  // USE_TDM

// ---------------------------------------------------------------------------
// Kernel 0: f32 -> bf16 conversion
// ---------------------------------------------------------------------------
__global__ __launch_bounds__(256)
void cvt_bf16_kernel(const float* __restrict__ src, __bf16* __restrict__ dst,
                     int n) {
  int i = blockIdx.x * 256 + threadIdx.x;
  if (i < n) dst[i] = (__bf16)src[i];
}

// ---------------------------------------------------------------------------
// Kernel 1: fused QKV projection + bias + RoPE.
//   xb:   [B*S, D]   bf16 (row-major)
//   wqkv: [3D, D]    bf16 (row-major, Wq|Wk|Wv stacked; C = X * W^T)
//   out:  q/k/v bufs [B][H][S][HD] bf16, RoPE applied to Q and K.
// grid = (3D/64, B*S/16), block = 128 (4 waves, one 16x16 tile each)
// ---------------------------------------------------------------------------
__global__ __launch_bounds__(128)
void qkv_rope_kernel(const __bf16* __restrict__ xb,
                     const __bf16* __restrict__ wqkv,
                     const float* __restrict__ bq,
                     const float* __restrict__ bk,
                     const float* __restrict__ bv,
                     __bf16* __restrict__ qbuf,
                     __bf16* __restrict__ kbuf,
                     __bf16* __restrict__ vbuf) {
  const int lane = threadIdx.x & 31;
  const int wv   = threadIdx.x >> 5;
  const int lm   = lane & 15;
  const int half = lane >> 4;
  const int n0 = blockIdx.x * 64 + wv * 16;
  const int m0 = blockIdx.y * 16;

  const __bf16* arow = xb   + (size_t)(m0 + lm) * D_;
  const __bf16* brow = wqkv + (size_t)(n0 + lm) * D_;

  v8f acc = {0.f, 0.f, 0.f, 0.f, 0.f, 0.f, 0.f, 0.f};
#pragma unroll 4
  for (int kk = 0; kk < D_; kk += 32) {
    __builtin_prefetch(arow + kk + 256, 0, 0);
    __builtin_prefetch(brow + kk + 256, 0, 0);
    v16bf a = load_a_frag(arow + kk, half);
    // B-fragment (32x16, K x N): lane col = lane%16, K-half by lane-half.
    v16bf b = *reinterpret_cast<const v16bf*>(brow + kk + half * 16);
    acc = WMMA_BF16(a, b, acc);
  }

  const int n = n0 + lm;
  const float bias = (n < D_) ? bq[n] : (n < 2 * D_ ? bk[n - D_] : bv[n - 2 * D_]);
  const bool  isv  = (n0 >= 2 * D_);             // uniform per wave
  const int   d    = n & (HD_ - 1);
  const int   h    = (n & (D_ - 1)) >> 6;
  const float invf = __powf(10000.0f, -(float)(d & ~1) * (1.0f / (float)HD_));
  __bf16* dstbuf = (n0 < D_) ? qbuf : (isv ? vbuf : kbuf);

#pragma unroll
  for (int i = 0; i < 8; ++i) {
    const int m    = m0 + i + half * 8;          // global row in [0, B*S)
    const int bidx = m >> 11;                    // batch
    const int spos = m & (S_ - 1);               // sequence position
    float val = acc[i] + bias;
    if (!isv) {                                  // RoPE on Q and K
      float partner = __shfl_xor(val, 1, 32);    // pair element (d ^ 1)
      float sn, cs;
      __sincosf((float)spos * invf, &sn, &cs);
      val = (d & 1) ? (partner * sn + val * cs)  // o1 = x0*s + x1*c
                    : (val * cs - partner * sn); // o0 = x0*c - x1*s
    }
    dstbuf[(((size_t)(bidx * H_ + h) * S_ + spos) * HD_) + d] = (__bf16)val;
  }
}

// ---------------------------------------------------------------------------
// Kernel 2: flash attention (causal + left-pad mask), online softmax.
// grid = (S/64, H, B), block = 128 (4 waves; each wave owns a 16-row q tile).
// K chunk (32x64) staged via the Tensor Data Mover; V^T staged manually.
// ---------------------------------------------------------------------------
__global__ __launch_bounds__(128)
void flash_attn_kernel(const __bf16* __restrict__ qbuf,
                       const __bf16* __restrict__ kbuf,
                       const __bf16* __restrict__ vbuf,
                       const int* __restrict__ eff,
                       __bf16* __restrict__ attnb) {
  __shared__ alignas(32) __bf16 Kt[32 * 64];      // K chunk  [key][hd]
  __shared__ alignas(32) __bf16 Vt[64 * 32];      // V^T      [hd][key]
  __shared__ alignas(32) __bf16 Pb[4][16 * 32];   // P tiles  [qrow][key]

  const int lane = threadIdx.x & 31;
  const int wv   = threadIdx.x >> 5;
  const int lm   = lane & 15;
  const int half = lane >> 4;
  const int b  = blockIdx.z;
  const int h  = blockIdx.y;
  const int qb0   = blockIdx.x * 64;
  const int qbase = qb0 + wv * 16;
  const int pad_limit = S_ - eff[b];              // cols >= pad_limit masked

  const __bf16* Kh = kbuf + (size_t)(b * H_ + h) * S_ * HD_;
  const __bf16* Vh = vbuf + (size_t)(b * H_ + h) * S_ * HD_;
  const __bf16* Qrow = qbuf + ((size_t)(b * H_ + h) * S_ + qbase + lm) * HD_;

  const v16bf qa0 = load_a_frag(Qrow, half);      // hd 0..31
  const v16bf qa1 = load_a_frag(Qrow + 32, half); // hd 32..63

  const v8f zero8 = {0.f, 0.f, 0.f, 0.f, 0.f, 0.f, 0.f, 0.f};
  v8f o[4];
  o[0] = zero8; o[1] = zero8; o[2] = zero8; o[3] = zero8;
  float mrow[8], lrow[8];
#pragma unroll
  for (int i = 0; i < 8; ++i) { mrow[i] = -3.0e38f; lrow[i] = 0.f; }

#ifdef USE_TDM
  const unsigned kt_lds_off = (unsigned)(size_t)(&Kt[0]);  // LDS byte offset
#endif

  const int kend = qb0 + 64;                      // causal bound for block
  for (int kb = 0; kb < kend; kb += 32) {
    // ---- staging --------------------------------------------------------
#ifdef USE_TDM
    if (wv == 0) {                                // one wave drives the TDM
      tdm_load_k_chunk(Kh + (size_t)kb * HD_, kt_lds_off);
    }
#endif
    {
      const int kr = threadIdx.x >> 2;            // 0..31 (key row)
      const int c0 = (threadIdx.x & 3) * 16;      // hd column base
#ifndef USE_TDM
      const v8bf* ks = reinterpret_cast<const v8bf*>(Kh + (size_t)(kb + kr) * HD_ + c0);
      v8bf k0 = ks[0], k1 = ks[1];
      *reinterpret_cast<v8bf*>(Kt + kr * 64 + c0)     = k0;
      *reinterpret_cast<v8bf*>(Kt + kr * 64 + c0 + 8) = k1;
#endif
      const v8bf* vs = reinterpret_cast<const v8bf*>(Vh + (size_t)(kb + kr) * HD_ + c0);
      v8bf v0 = vs[0], v1 = vs[1];
#pragma unroll
      for (int j = 0; j < 8; ++j) Vt[(c0 + j) * 32 + kr] = v0[j];
#pragma unroll
      for (int j = 0; j < 8; ++j) Vt[(c0 + 8 + j) * 32 + kr] = v1[j];
    }
#ifdef USE_TDM
    if (wv == 0) {
      __builtin_amdgcn_s_wait_tensorcnt(0);       // TDM chunk landed in LDS
    }
#endif
    __syncthreads();

    if (kb <= qbase + 15) {                       // wave-level causal skip
      // ---- S = Q * K^T for two 16-key subtiles -------------------------
      v8f s0 = zero8, s1 = zero8;
      {
        v16bf bk0a = *reinterpret_cast<const v16bf*>(Kt + lm * 64 + half * 16);
        v16bf bk0b = *reinterpret_cast<const v16bf*>(Kt + lm * 64 + 32 + half * 16);
        v16bf bk1a = *reinterpret_cast<const v16bf*>(Kt + (16 + lm) * 64 + half * 16);
        v16bf bk1b = *reinterpret_cast<const v16bf*>(Kt + (16 + lm) * 64 + 32 + half * 16);
        s0 = WMMA_BF16(qa0, bk0a, s0);
        s0 = WMMA_BF16(qa1, bk0b, s0);
        s1 = WMMA_BF16(qa0, bk1a, s1);
        s1 = WMMA_BF16(qa1, bk1b, s1);
      }
      // ---- mask + online softmax (row spans 16 lanes of one half) ------
      const int key0 = kb + lm;
      const int key1 = kb + 16 + lm;
#pragma unroll
      for (int i = 0; i < 8; ++i) {
        const int m = qbase + i + half * 8;
        float a0 = (key0 > m || key0 >= pad_limit) ? -3.0e38f : s0[i] * 0.125f;
        float a1 = (key1 > m || key1 >= pad_limit) ? -3.0e38f : s1[i] * 0.125f;
        float mx = fmaxf(a0, a1);
        mx = fmaxf(mx, __shfl_xor(mx, 1, 32));
        mx = fmaxf(mx, __shfl_xor(mx, 2, 32));
        mx = fmaxf(mx, __shfl_xor(mx, 4, 32));
        mx = fmaxf(mx, __shfl_xor(mx, 8, 32));
        const float nm    = fmaxf(mrow[i], mx);
        const float alpha = __expf(mrow[i] - nm);
        const float p0    = __expf(a0 - nm);
        const float p1    = __expf(a1 - nm);
        float ps = p0 + p1;
        ps += __shfl_xor(ps, 1, 32);
        ps += __shfl_xor(ps, 2, 32);
        ps += __shfl_xor(ps, 4, 32);
        ps += __shfl_xor(ps, 8, 32);
        lrow[i] = lrow[i] * alpha + ps;
        mrow[i] = nm;
        o[0][i] *= alpha; o[1][i] *= alpha; o[2][i] *= alpha; o[3][i] *= alpha;
        Pb[wv][(i + half * 8) * 32 + lm]      = (__bf16)p0;   // C-layout -> LDS
        Pb[wv][(i + half * 8) * 32 + 16 + lm] = (__bf16)p1;
      }
      // ---- O += P * V (P re-read from LDS in A-fragment layout) --------
      v16bf pa = load_a_frag(&Pb[wv][lm * 32], half);
#pragma unroll
      for (int nt = 0; nt < 4; ++nt) {
        v16bf vbf = *reinterpret_cast<const v16bf*>(Vt + (nt * 16 + lm) * 32 + half * 16);
        o[nt] = WMMA_BF16(pa, vbf, o[nt]);
      }
    }
    __syncthreads();
  }

  // ---- epilogue: normalize and write [B][S][D] bf16 ----------------------
#pragma unroll
  for (int i = 0; i < 8; ++i) {
    const float invl = 1.0f / lrow[i];
    const int m = qbase + i + half * 8;
    __bf16* dst = attnb + ((size_t)(b * S_ + m) * D_) + h * HD_;
#pragma unroll
    for (int nt = 0; nt < 4; ++nt)
      dst[nt * 16 + lm] = (__bf16)(o[nt][i] * invl);
  }
}

// ---------------------------------------------------------------------------
// Kernel 3: output projection, f32 result.  out = attn @ Wo^T + bo
// grid = (D/64, B*S/16), block = 128.
// ---------------------------------------------------------------------------
__global__ __launch_bounds__(128)
void oproj_kernel(const __bf16* __restrict__ attnb,
                  const __bf16* __restrict__ wob,
                  const float* __restrict__ bo,
                  float* __restrict__ out) {
  const int lane = threadIdx.x & 31;
  const int wv   = threadIdx.x >> 5;
  const int lm   = lane & 15;
  const int half = lane >> 4;
  const int n0 = blockIdx.x * 64 + wv * 16;
  const int m0 = blockIdx.y * 16;

  const __bf16* arow = attnb + (size_t)(m0 + lm) * D_;
  const __bf16* brow = wob   + (size_t)(n0 + lm) * D_;

  v8f acc = {0.f, 0.f, 0.f, 0.f, 0.f, 0.f, 0.f, 0.f};
#pragma unroll 4
  for (int kk = 0; kk < D_; kk += 32) {
    __builtin_prefetch(arow + kk + 256, 0, 0);
    __builtin_prefetch(brow + kk + 256, 0, 0);
    v16bf a = load_a_frag(arow + kk, half);
    v16bf b = *reinterpret_cast<const v16bf*>(brow + kk + half * 16);
    acc = WMMA_BF16(a, b, acc);
  }

  const int n = n0 + lm;
  const float bias = bo[n];
#pragma unroll
  for (int i = 0; i < 8; ++i) {
    const int m = m0 + i + half * 8;
    out[(size_t)m * D_ + n] = acc[i] + bias;
  }
}

// ---------------------------------------------------------------------------
// Host launch
// ---------------------------------------------------------------------------
extern "C" void kernel_launch(void* const* d_in, const int* in_sizes, int n_in,
                              void* d_out, int out_size, void* d_ws,
                              size_t ws_size, hipStream_t stream) {
  const float* x  = (const float*)d_in[0];
  const int*   el = (const int*)d_in[1];
  const float* Wq = (const float*)d_in[2];
  const float* bq = (const float*)d_in[3];
  const float* Wk = (const float*)d_in[4];
  const float* bk = (const float*)d_in[5];
  const float* Wv = (const float*)d_in[6];
  const float* bv = (const float*)d_in[7];
  const float* Wo = (const float*)d_in[8];
  const float* bo = (const float*)d_in[9];
  float* out = (float*)d_out;

  const size_t MROWS = (size_t)B_ * S_;        // 4096
  const size_t XEL   = MROWS * D_;             // 4 Mi elems
  const size_t WEL   = (size_t)D_ * D_;        // 1 Mi elems
  const size_t QKVEL = (size_t)B_ * H_ * S_ * HD_; // 4 Mi elems

  __bf16* ws    = (__bf16*)d_ws;
  __bf16* xb    = ws;                          // [B*S, D]
  __bf16* wqkv  = xb + XEL;                    // [3D, D]
  __bf16* wob   = wqkv + 3 * WEL;              // [D, D]
  __bf16* qbuf  = wob + WEL;                   // [B][H][S][HD]
  __bf16* kbuf  = qbuf + QKVEL;
  __bf16* vbuf  = kbuf + QKVEL;
  __bf16* attnb = vbuf + QKVEL;                // [B*S, D]

  // f32 -> bf16 conversions
  cvt_bf16_kernel<<<dim3((unsigned)((XEL + 255) / 256)), 256, 0, stream>>>(x, xb, (int)XEL);
  cvt_bf16_kernel<<<dim3((unsigned)((WEL + 255) / 256)), 256, 0, stream>>>(Wq, wqkv, (int)WEL);
  cvt_bf16_kernel<<<dim3((unsigned)((WEL + 255) / 256)), 256, 0, stream>>>(Wk, wqkv + WEL, (int)WEL);
  cvt_bf16_kernel<<<dim3((unsigned)((WEL + 255) / 256)), 256, 0, stream>>>(Wv, wqkv + 2 * WEL, (int)WEL);
  cvt_bf16_kernel<<<dim3((unsigned)((WEL + 255) / 256)), 256, 0, stream>>>(Wo, wob, (int)WEL);

  // QKV projection + RoPE
  qkv_rope_kernel<<<dim3(3 * D_ / 64, (unsigned)(MROWS / 16)), 128, 0, stream>>>(
      xb, wqkv, bq, bk, bv, qbuf, kbuf, vbuf);

  // Flash attention
  flash_attn_kernel<<<dim3(S_ / 64, H_, B_), 128, 0, stream>>>(
      qbuf, kbuf, vbuf, el, attnb);

  // Output projection
  oproj_kernel<<<dim3(D_ / 64, (unsigned)(MROWS / 16)), 128, 0, stream>>>(
      attnb, wob, bo, out);

  (void)in_sizes; (void)n_in; (void)out_size; (void)ws_size;
}
